// MHA_32581621908149
// MI455X (gfx1250) — matrix operand order repaired and checked by reading, not compiled, not run on hardware
//
#include <hip/hip_runtime.h>
#include <cstdint>

// ---------------------------------------------------------------------------
// Causal MHA forward for MI455X (gfx1250, wave32, WMMA f16->f32).
// Pipeline: f32->f16 convert, QKV GEMM (async-LDS staged WMMA) -> Q/K/Vt,
// flash attention (WMMA QK^T + online softmax + WMMA PV), out-proj GEMM.
// ---------------------------------------------------------------------------

typedef _Float16 half_t;
typedef __attribute__((ext_vector_type(8)))  _Float16 v8h;
typedef __attribute__((ext_vector_type(16))) _Float16 v16h;
typedef __attribute__((ext_vector_type(8)))  float    v8f;

#define WMMA_F16(a, b, c) \
  __builtin_amdgcn_wmma_f32_16x16x32_f16(false, (a), false, (b), (short)0, (c), false, false)

__device__ __forceinline__ v16h combine16(v8h lo, v8h hi) {
  return __builtin_shufflevector(lo, hi, 0, 1, 2, 3, 4, 5, 6, 7, 8, 9, 10, 11, 12, 13, 14, 15);
}

// A fragment: 16x32 f16 tile, element (m,k) at src[m*ld + k].
// ISA 7.12.2: lanes 0-15 hold row m=lane, K in {0..7,16..23}; lanes 16-31 K in {8..15,24..31}.
__device__ __forceinline__ v16h load_frag_a(const half_t* src, int ld, int lane) {
  const half_t* row = src + (size_t)(lane & 15) * ld + ((lane >> 4) ? 8 : 0);
  v8h lo = *(const v8h*)(row);
  v8h hi = *(const v8h*)(row + 16);
  return combine16(lo, hi);
}

// B fragment: 32x16, element (k,n) at src[n*ld + k] (source stores B^T rows).
// Lanes 0-15: n=lane, K in {0..7,16..23}; lanes 16-31: K in {8..15,24..31}.
__device__ __forceinline__ v16h load_frag_b(const half_t* src, int ld, int lane) {
  const half_t* row = src + (size_t)(lane & 15) * ld + ((lane >> 4) ? 8 : 0);
  v8h lo = *(const v8h*)(row);
  v8h hi = *(const v8h*)(row + 16);
  return combine16(lo, hi);
}

// Async 16-byte global -> LDS copy (ASYNCcnt-tracked, no VGPR data path).
// VGLOBAL async encoding: VDST = LDS byte address, VADDR = 64-bit global addr.
__device__ __forceinline__ void async_copy_b128(const half_t* lds_ptr, const half_t* gptr) {
  unsigned lds = (unsigned)(uintptr_t)lds_ptr;           // addr[31:0] = LDS offset
  unsigned long long ga = (unsigned long long)(uintptr_t)gptr;
  asm volatile("global_load_async_to_lds_b128 %0, %1, off"
               :: "v"(lds), "v"(ga) : "memory");
}

// ---------------------------------------------------------------------------
// Block-tiled NT GEMM core: C(64x128) += A(64xK) * B^T(128xK).
// 8 waves as 2(M) x 4(N); each wave owns a 32x32 sub-tile (2x2 WMMA frags).
// A/B K-slices staged into LDS with async b128 copies each BK=64 step.
// ---------------------------------------------------------------------------
__device__ __forceinline__ void gemm_block_tile(const half_t* __restrict__ Ag,
                                                const half_t* __restrict__ Bg,
                                                int Kdim, int m0, int n0,
                                                half_t* As, half_t* Bs,
                                                v8f c[2][2]) {
  const int tid  = threadIdx.x;
  const int lane = tid & 31;
  const int wave = tid >> 5;
  const int wm   = (wave >> 2) * 32;      // 0 or 32
  const int wn   = (wave & 3) * 32;       // 0,32,64,96

  c[0][0] = (v8f){}; c[0][1] = (v8f){}; c[1][0] = (v8f){}; c[1][1] = (v8f){};

  for (int kk = 0; kk < Kdim; kk += 64) {
    // Stage A tile: 64 rows x 64 halves = 512 b128 chunks (2 per thread)
#pragma unroll
    for (int i = 0; i < 2; ++i) {
      const int idx = tid + i * 256;
      const int row = idx >> 3, seg = idx & 7;
      async_copy_b128(As + row * 64 + seg * 8,
                      Ag + (size_t)(m0 + row) * Kdim + kk + seg * 8);
    }
    // Stage B tile: 128 rows x 64 halves = 1024 b128 chunks (4 per thread)
#pragma unroll
    for (int i = 0; i < 4; ++i) {
      const int idx = tid + i * 256;
      const int row = idx >> 3, seg = idx & 7;
      async_copy_b128(Bs + row * 64 + seg * 8,
                      Bg + (size_t)(n0 + row) * Kdim + kk + seg * 8);
    }
    asm volatile("s_wait_asynccnt 0x0" ::: "memory");
    __syncthreads();

#pragma unroll
    for (int ks = 0; ks < 64; ks += 32) {
      v16h a0 = load_frag_a(As + (wm + 0)  * 64 + ks, 64, lane);
      v16h a1 = load_frag_a(As + (wm + 16) * 64 + ks, 64, lane);
      v16h b0 = load_frag_b(Bs + (wn + 0)  * 64 + ks, 64, lane);
      v16h b1 = load_frag_b(Bs + (wn + 16) * 64 + ks, 64, lane);
      c[0][0] = WMMA_F16(a0, b0, c[0][0]);
      c[0][1] = WMMA_F16(a0, b1, c[0][1]);
      c[1][0] = WMMA_F16(a1, b0, c[1][0]);
      c[1][1] = WMMA_F16(a1, b1, c[1][1]);
    }
    __syncthreads();   // all waves done reading LDS before next stage
  }
}

// --------------------------- f32 -> f16 convert ----------------------------
__global__ void __launch_bounds__(256)
cvt_f32_f16_kernel(const float* __restrict__ src, half_t* __restrict__ dst, int n) {
  int i = blockIdx.x * 256 + threadIdx.x;
  if (i < n) dst[i] = (half_t)src[i];
}

// ------------------------------- QKV GEMM ----------------------------------
// Xh: [4096 x 2048] f16, Wh: [6144 x 2048] f16 (torch Linear layout, NT GEMM).
// Scatter epilogue -> Q[bh][s][128], K[bh][s][128], Vt[bh][128][s] (bh=b*16+h).
__global__ void __launch_bounds__(256)
gemm_qkv_kernel(const half_t* __restrict__ Xh, const half_t* __restrict__ Wh,
                half_t* __restrict__ Q, half_t* __restrict__ K,
                half_t* __restrict__ Vt) {
  __shared__ __align__(16) half_t As[64 * 64];
  __shared__ __align__(16) half_t Bs[128 * 64];
  const int tM = blockIdx.x & 63;   // 4096/64 = 64 M-tiles
  const int tN = blockIdx.x >> 6;   // 6144/128 = 48 N-tiles

  v8f c[2][2];
  gemm_block_tile(Xh, Wh, 2048, tM * 64, tN * 128, As, Bs, c);

  const int lane = threadIdx.x & 31;
  const int wave = threadIdx.x >> 5;
  const int wm = (wave >> 2) * 32, wn = (wave & 3) * 32;
  const int l15 = lane & 15, hg = lane >> 4;
#pragma unroll
  for (int fm = 0; fm < 2; ++fm) {
#pragma unroll
    for (int fn = 0; fn < 2; ++fn) {
      const int e = tN * 128 + wn + fn * 16 + l15;
      const int sel = e >> 11, eh = e & 2047, h = eh >> 7, d = eh & 127;
#pragma unroll
      for (int r = 0; r < 8; ++r) {
        const int t = tM * 64 + wm + fm * 16 + r + 8 * hg;
        const int b = t >> 11, s = t & 2047;
        const half_t val = (half_t)c[fm][fn][r];
        const size_t bh = (size_t)(b * 16 + h);
        if (sel == 0)      Q [(bh * 2048 + s) * 128 + d] = val;
        else if (sel == 1) K [(bh * 2048 + s) * 128 + d] = val;
        else               Vt[(bh * 128 + d) * 2048 + s] = val;
      }
    }
  }
}

// ---------------------------- Flash attention ------------------------------
// One wave per 16-row Q tile; K/V consumed 32 keys per step; online softmax.
__global__ void __launch_bounds__(128)
attn_kernel(const half_t* __restrict__ Q, const half_t* __restrict__ K,
            const half_t* __restrict__ Vt, half_t* __restrict__ Y) {
  const int lane = threadIdx.x & 31;
  const int wave = threadIdx.x >> 5;
  const int bh   = blockIdx.x;                // b*16 + h
  const int qt   = blockIdx.y * 4 + wave;     // 0..127
  const int q0   = qt * 16;
  const int l15  = lane & 15;
  const int hg   = lane >> 4;

  const half_t* Qp = Q  + (size_t)bh * 2048 * 128;
  const half_t* Kp = K  + (size_t)bh * 2048 * 128;
  const half_t* Vp = Vt + (size_t)bh * 128 * 2048;

  __shared__ __align__(16) half_t pbuf[4][16 * 32];
  half_t* pw = pbuf[wave];

  v16h qf[4];
#pragma unroll
  for (int c = 0; c < 4; ++c)
    qf[c] = load_frag_a(Qp + (size_t)q0 * 128 + c * 32, 128, lane);

  const v8f vzero = {};
  v8f o[8];
#pragma unroll
  for (int c = 0; c < 8; ++c) o[c] = vzero;
  float mrow[8], lrow[8];
#pragma unroll
  for (int r = 0; r < 8; ++r) { mrow[r] = -3.0e38f; lrow[r] = 0.0f; }

  const float scale = 0.08838834764831845f;   // 1/sqrt(128)
  const int kt_end = (q0 + 16 + 31) >> 5;     // causal: keys 0 .. q0+15

  for (int kt = 0; kt < kt_end; ++kt) {
    const int kbase = kt * 32;
    v8f s0 = vzero, s1 = vzero;
#pragma unroll
    for (int c = 0; c < 4; ++c) {
      v16h b0 = load_frag_b(Kp + (size_t)kbase * 128 + c * 32, 128, lane);
      v16h b1 = load_frag_b(Kp + (size_t)(kbase + 16) * 128 + c * 32, 128, lane);
      s0 = WMMA_F16(qf[c], b0, s0);
      s1 = WMMA_F16(qf[c], b1, s1);
    }
    const int col0 = kbase + l15, col1 = col0 + 16;
#pragma unroll
    for (int r = 0; r < 8; ++r) {
      const int qrow = q0 + r + 8 * hg;
      float v0 = (col0 <= qrow) ? (float)s0[r] * scale : -3.0e38f;
      float v1 = (col1 <= qrow) ? (float)s1[r] * scale : -3.0e38f;
      float m = fmaxf(v0, v1);
      m = fmaxf(m, __shfl_xor(m, 1, 32));
      m = fmaxf(m, __shfl_xor(m, 2, 32));
      m = fmaxf(m, __shfl_xor(m, 4, 32));
      m = fmaxf(m, __shfl_xor(m, 8, 32));
      const float mnew = fmaxf(mrow[r], m);
      const float fac  = __expf(mrow[r] - mnew);
      mrow[r] = mnew;
      const float p0 = __expf(v0 - mnew);
      const float p1 = __expf(v1 - mnew);
      float rs = p0 + p1;
      rs += __shfl_xor(rs, 1, 32);
      rs += __shfl_xor(rs, 2, 32);
      rs += __shfl_xor(rs, 4, 32);
      rs += __shfl_xor(rs, 8, 32);
      lrow[r] = lrow[r] * fac + rs;
#pragma unroll
      for (int c = 0; c < 8; ++c) o[c][r] *= fac;
      s0[r] = p0; s1[r] = p1;
    }
    // Re-pack P from C-layout to A-layout via LDS (same-wave producer/consumer)
#pragma unroll
    for (int r = 0; r < 8; ++r) {
      const int row = r + 8 * hg;
      pw[row * 32 + l15]      = (half_t)s0[r];
      pw[row * 32 + 16 + l15] = (half_t)s1[r];
    }
    asm volatile("s_wait_dscnt 0x0" ::: "memory");
    v16h pa = load_frag_a(pw, 32, lane);
#pragma unroll
    for (int c = 0; c < 8; ++c) {
      v16h vb = load_frag_b(Vp + (size_t)c * 16 * 2048 + kbase, 2048, lane);
      o[c] = WMMA_F16(pa, vb, o[c]);
    }
  }

  // Normalize and write y[t][h*128 + col] (f16, row-major [4096 x 2048])
  const size_t trow = (size_t)((bh >> 4) * 2048 + q0);
  const int h = bh & 15;
#pragma unroll
  for (int r = 0; r < 8; ++r) {
    const float inv = 1.0f / lrow[r];
    const int row = r + 8 * hg;
    half_t* yrow = Y + (trow + row) * 2048 + h * 128 + l15;
#pragma unroll
    for (int c = 0; c < 8; ++c)
      yrow[c * 16] = (half_t)((float)o[c][r] * inv);
  }
}

// --------------------------- Output projection -----------------------------
__global__ void __launch_bounds__(256)
gemm_out_kernel(const half_t* __restrict__ Yh, const half_t* __restrict__ Woh,
                float* __restrict__ Out) {
  __shared__ __align__(16) half_t As[64 * 64];
  __shared__ __align__(16) half_t Bs[128 * 64];
  const int tM = blockIdx.x & 63;   // 4096/64 = 64
  const int tN = blockIdx.x >> 6;   // 2048/128 = 16

  v8f c[2][2];
  gemm_block_tile(Yh, Woh, 2048, tM * 64, tN * 128, As, Bs, c);

  const int lane = threadIdx.x & 31;
  const int wave = threadIdx.x >> 5;
  const int wm = (wave >> 2) * 32, wn = (wave & 3) * 32;
  const int l15 = lane & 15, hg = lane >> 4;
#pragma unroll
  for (int fm = 0; fm < 2; ++fm) {
#pragma unroll
    for (int fn = 0; fn < 2; ++fn) {
#pragma unroll
      for (int r = 0; r < 8; ++r) {
        const int t = tM * 64 + wm + fm * 16 + r + 8 * hg;
        const int col = tN * 128 + wn + fn * 16 + l15;
        Out[(size_t)t * 2048 + col] = c[fm][fn][r];
      }
    }
  }
}

// ------------------------------- launcher ----------------------------------
extern "C" void kernel_launch(void* const* d_in, const int* in_sizes, int n_in,
                              void* d_out, int out_size, void* d_ws, size_t ws_size,
                              hipStream_t stream) {
  (void)in_sizes; (void)n_in; (void)out_size; (void)ws_size;
  constexpr int TOK = 2 * 2048;      // 4096 tokens
  constexpr int D   = 2048;
  constexpr int QKV = 3 * D;         // 6144

  const float* x    = (const float*)d_in[0];
  const float* Wqkv = (const float*)d_in[1];
  const float* Wo   = (const float*)d_in[2];
  float* out        = (float*)d_out;

  char* w = (char*)d_ws;
  auto take = [&](size_t bytes) {
    char* r = w;
    w += (bytes + 255) & ~(size_t)255;
    return r;
  };
  half_t* Xh   = (half_t*)take((size_t)TOK * D * 2);
  half_t* Wqh  = (half_t*)take((size_t)QKV * D * 2);
  half_t* Woh  = (half_t*)take((size_t)D * D * 2);
  half_t* Qb   = (half_t*)take((size_t)TOK * D * 2);   // [b*16+h][s][128]
  half_t* Kb   = (half_t*)take((size_t)TOK * D * 2);   // [b*16+h][s][128]
  half_t* Vtb  = (half_t*)take((size_t)TOK * D * 2);   // [b*16+h][128][s]
  half_t* Yh   = (half_t*)take((size_t)TOK * D * 2);   // [t][2048]

  {
    int n = TOK * D;
    cvt_f32_f16_kernel<<<(n + 255) / 256, 256, 0, stream>>>(x, Xh, n);
    n = QKV * D;
    cvt_f32_f16_kernel<<<(n + 255) / 256, 256, 0, stream>>>(Wqkv, Wqh, n);
    n = D * D;
    cvt_f32_f16_kernel<<<(n + 255) / 256, 256, 0, stream>>>(Wo, Woh, n);
  }

  // QKV: 64 x 48 block tiles of 64x128
  gemm_qkv_kernel<<<64 * 48, 256, 0, stream>>>(Xh, Wqh, Qb, Kb, Vtb);

  // Attention: grid (32 heads*batch, 128 qtiles / 4 waves)
  attn_kernel<<<dim3(32, 32), 128, 0, stream>>>(Qb, Kb, Vtb, Yh);

  // Out-proj: 64 x 16 block tiles of 64x128
  gemm_out_kernel<<<64 * 16, 256, 0, stream>>>(Yh, Woh, out);
}
